// SmoothRankAP_27178553049115
// MI455X (gfx1250) — matrix operand order repaired and checked by reading, not compiled, not run on hardware
//
#include <hip/hip_runtime.h>

typedef __attribute__((ext_vector_type(16))) _Float16 v16h;
typedef __attribute__((ext_vector_type(8)))  float    v8f;

#define BDIM 512      // batch size per reference
#define NW   8        // waves per block (wave32)
#define NTHREADS 256

// gfx1250 has native V_TANH_F32 (TRANS). Prefer the builtin, fall back to asm.
__device__ __forceinline__ float fast_tanh_f32(float x) {
#if __has_builtin(__builtin_amdgcn_tanhf)
  return __builtin_amdgcn_tanhf(x);
#else
  float r;
  asm volatile("v_tanh_f32 %0, %1" : "=v"(r) : "v"(x));
  return r;
#endif
}

// One block per query b. Computes ap[b].
//   sg(b,i,j) = 0.5 + 0.5*tanh(h_j - h_i),  h = 50*scores[b,:]   (50 = 0.5/TAU)
//   only rows i with target[b,i]==1 contribute; mask m_j = target[b,j]
//   sim_all = 0.5*T_all + 0.5*B + 0.5
//   sim_pos = 0.5*T_pos + 0.5*cnt + 0.5
// T_all / T_pos come out of one v_wmma_f32_16x16x32_f16 per 16(i)x32(j) tile:
// B-matrix column 0 = ones, column 1 = mask, rest = 0.
__global__ __launch_bounds__(NTHREADS)
void smoothap_per_query(const float* __restrict__ scores,
                        const float* __restrict__ target,
                        float* __restrict__ ap_out) {
  __shared__ float h[BDIM];
  __shared__ float m[BDIM];
  __shared__ int   idxList[BDIM];
  __shared__ int   cnt_s;
  __shared__ float partAll[NW][16];
  __shared__ float partPos[NW][16];
  __shared__ float contribArr[16];
  __shared__ float apAcc;

  const int b    = blockIdx.x;
  const int tid  = threadIdx.x;
  const int wave = tid >> 5;
  const int lane = tid & 31;
  const int halfId = lane >> 4;   // which 16-lane half of the wave
  const int nrow   = lane & 15;   // A: row M within tile; B: column N

  // Stage scaled score row and positive mask row into LDS (2KB + 2KB)
  for (int j = tid; j < BDIM; j += NTHREADS) {
    h[j] = scores[b * BDIM + j] * 50.0f;   // 0.5 / TAU
    m[j] = target[b * BDIM + j];
  }
  __syncthreads();

  // Deterministic compact list of positive rows (includes i==b, so cnt >= 1)
  if (tid == 0) {
    int c = 0;
    for (int j = 0; j < BDIM; ++j)
      if (m[j] > 0.5f) idxList[c++] = j;
    cnt_s = c;
    apAcc = 0.0f;
  }
  __syncthreads();
  const int cnt    = cnt_s;
  const int nTiles = (cnt + 15) >> 4;

  // Fragment index bases (WMMA 16x16x32 f16 layouts, wave32)
  const int kbA = halfId * 8;    // A: K = kbA+e (e<8), kbA+16+e-8 (e>=8)
  const int kbB = halfId * 16;   // unused; B is built with A's kbA K order

  // --- Build the two B fragments for this wave's j-chunks, branch-free. ---
  // Column weights: N==0 -> ones column (T_all), N==1 -> mask column (T_pos).
  const float c0w = (nrow == 0) ? 1.0f : 0.0f;
  const float c1w = (nrow == 1) ? 1.0f : 0.0f;
  const int j0a = wave * 32;            // chunk 0 for this wave
  const int j0b = (wave + NW) * 32;     // chunk 1 for this wave
  v16h B0, B1;
#pragma unroll
  for (int e = 0; e < 8; ++e) {
    // unconditional LDS loads (contiguous -> ds_load_b128), arithmetic select
    B0[e]     = (_Float16)(c0w + c1w * m[j0a + kbA + e]);
    B0[e + 8] = (_Float16)(c0w + c1w * m[j0a + kbA + 16 + e]);
    B1[e]     = (_Float16)(c0w + c1w * m[j0b + kbA + e]);
    B1[e + 8] = (_Float16)(c0w + c1w * m[j0b + kbA + 16 + e]);
  }

  for (int tile = 0; tile < nTiles; ++tile) {
    // Row index this lane supplies for the A fragment (clamp-pad last tile)
    int p = tile * 16 + nrow;
    if (p >= cnt) p = cnt - 1;
    const float hi = h[idxList[p]];

    v8f acc = {0.f, 0.f, 0.f, 0.f, 0.f, 0.f, 0.f, 0.f};

    // ---- chunk 0 ----
    {
      v16h A;
#pragma unroll
      for (int e = 0; e < 8; ++e) {
        A[e]     = (_Float16)fast_tanh_f32(h[j0a + kbA + e]      - hi);
        A[e + 8] = (_Float16)fast_tanh_f32(h[j0a + kbA + 16 + e] - hi);
      }
      acc = __builtin_amdgcn_wmma_f32_16x16x32_f16(
          false, A, false, B0, (short)0, acc, false, false);
    }
    // ---- chunk 1 ----
    {
      v16h A;
#pragma unroll
      for (int e = 0; e < 8; ++e) {
        A[e]     = (_Float16)fast_tanh_f32(h[j0b + kbA + e]      - hi);
        A[e + 8] = (_Float16)fast_tanh_f32(h[j0b + kbA + 16 + e] - hi);
      }
      acc = __builtin_amdgcn_wmma_f32_16x16x32_f16(
          false, A, false, B1, (short)0, acc, false, false);
    }

    // Extract columns 0/1 of the f32 accumulator into fixed LDS slots.
    // Column 0 lives in lanes {0,16}, column 1 in lanes {1,17}.
    if (nrow < 2) {
      float* dst = (nrow == 0) ? &partAll[wave][0] : &partPos[wave][0];
      const int base = halfId * 8;   // lanes 16/17 hold rows 8..15
#pragma unroll
      for (int r = 0; r < 8; ++r) dst[base + r] = acc[r];
    }
    __syncthreads();

    // Deterministic cross-wave reduction + per-row AP contribution
    if (tid < 16) {
      float sA = 0.f, sP = 0.f;
#pragma unroll
      for (int w = 0; w < NW; ++w) { sA += partAll[w][tid]; sP += partPos[w][tid]; }
      float simPos = 0.5f * sP + 0.5f * (float)cnt  + 0.5f;
      float simAll = 0.5f * sA + 0.5f * (float)BDIM + 0.5f;
      contribArr[tid] = simPos / simAll;
    }
    __syncthreads();
    if (tid == 0) {
      int rmax = cnt - tile * 16;
      if (rmax > 16) rmax = 16;
      float s = apAcc;
      for (int r = 0; r < rmax; ++r) s += contribArr[r];
      apAcc = s;
    }
    __syncthreads();
  }

  if (tid == 0) ap_out[b] = apAcc / (float)cnt;
}

// Deterministic final reduction: 1 - mean(ap)
__global__ void smoothap_finalize(const float* __restrict__ ap,
                                  float* __restrict__ out) {
  if (blockIdx.x == 0 && threadIdx.x == 0) {
    float s = 0.0f;
    for (int b = 0; b < BDIM; ++b) s += ap[b];
    out[0] = 1.0f - s / (float)BDIM;
  }
}

extern "C" void kernel_launch(void* const* d_in, const int* in_sizes, int n_in,
                              void* d_out, int out_size, void* d_ws, size_t ws_size,
                              hipStream_t stream) {
  const float* scores = (const float*)d_in[0];   // [512,512] f32
  const float* target = (const float*)d_in[1];   // [512,512] f32
  float* ap = (float*)d_ws;                      // 512 floats of scratch
  (void)in_sizes; (void)n_in; (void)out_size; (void)ws_size;

  smoothap_per_query<<<BDIM, NTHREADS, 0, stream>>>(scores, target, ap);
  smoothap_finalize<<<1, 32, 0, stream>>>(ap, (float*)d_out);
}
